// GINEncoder_68813966016847
// MI455X (gfx1250) — matrix operand order repaired and checked by reading, not compiled, not run on hardware
//
#include <hip/hip_runtime.h>
#include <hip/hip_bf16.h>

// ---------------------------------------------------------------------------
// GIN encoder for MI455X (gfx1250, wave32, WMMA bf16 with fp32 accumulate)
// ---------------------------------------------------------------------------

typedef __attribute__((ext_vector_type(16))) __bf16 v16bf;
typedef __attribute__((ext_vector_type(8)))  float  v8f;

union FragBF {
    v16bf v;
    uint4 q[2];
};

#define DHID 256

__device__ __forceinline__ unsigned short f2bf(float f) {
    unsigned int u = __float_as_uint(f);
    unsigned int r = u + 0x7FFFu + ((u >> 16) & 1u);   // round-to-nearest-even
    return (unsigned short)(r >> 16);
}

__device__ __forceinline__ void atomic_add_f32(float* p, float v) {
    __hip_atomic_fetch_add(p, v, __ATOMIC_RELAXED, __HIP_MEMORY_SCOPE_AGENT);
}

// --------------------------- zero fill -------------------------------------
__global__ __launch_bounds__(256) void zero_kernel(float* __restrict__ p, long long n) {
    long long i = (long long)blockIdx.x * 256 + threadIdx.x;
    if (i < n) p[i] = 0.0f;
}

// ------------------- weight convert: W[KxNc] fp32 -> Wt[n*K+k] bf16 --------
__global__ __launch_bounds__(256) void convert_w_kernel(const float* __restrict__ W,
                                                        unsigned short* __restrict__ Wt,
                                                        int K, int Nc) {
    int idx = blockIdx.x * 256 + threadIdx.x;
    if (idx >= K * Nc) return;
    int k = idx / Nc;
    int n = idx - k * Nc;
    Wt[(size_t)n * K + k] = f2bf(W[idx]);   // coalesced read, transposed write
}

// ------------------- edge scatter: agg[dst] += h[src] ----------------------
__global__ __launch_bounds__(256) void scatter_kernel(const float* __restrict__ h,
                                                      const int* __restrict__ src,
                                                      const int* __restrict__ dst,
                                                      float* __restrict__ agg,
                                                      int E, int D) {
    long long t = (long long)blockIdx.x * 256 + threadIdx.x;
    int perE = D >> 2;                         // float4 chunks per edge
    long long total = (long long)E * perE;
    if (t >= total) return;
    int e = (int)(t / perE);
    int f = ((int)(t - (long long)e * perE)) << 2;
    int s = src[e];
    int d = dst[e];
    const float4 v = *(const float4*)&h[(size_t)s * D + f];
    float* o = &agg[(size_t)d * D + f];
    atomic_add_f32(o + 0, v.x);
    atomic_add_f32(o + 1, v.y);
    atomic_add_f32(o + 2, v.z);
    atomic_add_f32(o + 3, v.w);
}

// ---------------------------------------------------------------------------
// Fused GIN MLP:  out = relu( relu((hin+agg) @ Wa + ba) @ Wb + bb )
// 16-row block per workgroup, 8 waves, each wave owns two 16-col tiles.
// Weights are bf16, N-major (Wt[n*K + k]) so B fragments are contiguous.
// ---------------------------------------------------------------------------
__global__ __launch_bounds__(256)
void gin_mlp_kernel(const float* hin,                 // may alias hout (disjoint-row safe)
                    const float* __restrict__ agg,
                    const unsigned short* __restrict__ WaT, const float* __restrict__ ba,
                    const unsigned short* __restrict__ WbT, const float* __restrict__ bb,
                    float* hout, int N, int Kin) {
    __shared__ unsigned short ldsA[16 * 256];     // staged (hin+agg) block, bf16
    __shared__ unsigned short ldsT[16 * DHID];    // hidden activations, bf16

    const int m0  = blockIdx.x * 16;
    const int tid = threadIdx.x;

    // ---- stage A = hin + agg into LDS as bf16 (uniform trip count) ----
    for (int idx = tid; idx < 16 * Kin; idx += 256) {
        int r = idx / Kin;
        int c = idx - r * Kin;
        int gm = m0 + r;
        float v = 0.0f;
        if (gm < N) {
            size_t o = (size_t)gm * Kin + c;
            v = hin[o] + agg[o];
        }
        ldsA[r * Kin + c] = f2bf(v);
    }
    __syncthreads();

    const int wid  = tid >> 5;
    const int lane = tid & 31;
    const int half = lane >> 4;      // 0: lanes 0-15, 1: lanes 16-31
    const int nl   = lane & 15;
    const int n0   = wid * 32;       // first column tile of this wave
    const int n1   = n0 + 16;        // second column tile

    // -------------------- GEMM1: T = relu(A @ Wa + ba) --------------------
    {
        v8f acc0 = {0, 0, 0, 0, 0, 0, 0, 0};
        v8f acc1 = {0, 0, 0, 0, 0, 0, 0, 0};
        const float bias0 = ba[n0 + nl];
        const float bias1 = ba[n1 + nl];
        const unsigned short* pb0 = &WaT[(size_t)(n0 + nl) * Kin + half * 8];
        const unsigned short* pb1 = &WaT[(size_t)(n1 + nl) * Kin + half * 8];
        for (int kb = 0; kb < Kin; kb += 32) {
            FragBF a, b0, b1;
            const unsigned short* pa = &ldsA[nl * Kin + kb + half * 8];
            a.q[0] = *(const uint4*)(pa);        // K = kb+half*8 .. +7
            a.q[1] = *(const uint4*)(pa + 16);   // K = kb+16+half*8 .. +7
            b0.q[0] = *(const uint4*)(pb0 + kb);
            b0.q[1] = *(const uint4*)(pb0 + kb + 16);
            b1.q[0] = *(const uint4*)(pb1 + kb);
            b1.q[1] = *(const uint4*)(pb1 + kb + 16);
            __builtin_prefetch((const void*)(pb0 + kb + 32), 0, 1);
            __builtin_prefetch((const void*)(pb1 + kb + 32), 0, 1);
            acc0 = __builtin_amdgcn_wmma_f32_16x16x32_bf16(false, a.v, false, b0.v,
                                                           (short)0, acc0, false, false);
            acc1 = __builtin_amdgcn_wmma_f32_16x16x32_bf16(false, a.v, false, b1.v,
                                                           (short)0, acc1, false, false);
        }
#pragma unroll
        for (int r = 0; r < 8; ++r) {            // C layout: lane = column, vgpr r = row
            int m = half * 8 + r;
            float v0 = acc0[r] + bias0; v0 = v0 > 0.0f ? v0 : 0.0f;
            float v1 = acc1[r] + bias1; v1 = v1 > 0.0f ? v1 : 0.0f;
            ldsT[m * DHID + n0 + nl] = f2bf(v0);
            ldsT[m * DHID + n1 + nl] = f2bf(v1);
        }
    }
    __syncthreads();

    // -------------------- GEMM2: out = relu(T @ Wb + bb) -------------------
    {
        v8f acc0 = {0, 0, 0, 0, 0, 0, 0, 0};
        v8f acc1 = {0, 0, 0, 0, 0, 0, 0, 0};
        const float bias0 = bb[n0 + nl];
        const float bias1 = bb[n1 + nl];
        const unsigned short* pb0 = &WbT[(size_t)(n0 + nl) * DHID + half * 8];
        const unsigned short* pb1 = &WbT[(size_t)(n1 + nl) * DHID + half * 8];
        for (int kb = 0; kb < DHID; kb += 32) {
            FragBF a, b0, b1;
            const unsigned short* pa = &ldsT[nl * DHID + kb + half * 8];
            a.q[0] = *(const uint4*)(pa);
            a.q[1] = *(const uint4*)(pa + 16);
            b0.q[0] = *(const uint4*)(pb0 + kb);
            b0.q[1] = *(const uint4*)(pb0 + kb + 16);
            b1.q[0] = *(const uint4*)(pb1 + kb);
            b1.q[1] = *(const uint4*)(pb1 + kb + 16);
            __builtin_prefetch((const void*)(pb0 + kb + 32), 0, 1);
            __builtin_prefetch((const void*)(pb1 + kb + 32), 0, 1);
            acc0 = __builtin_amdgcn_wmma_f32_16x16x32_bf16(false, a.v, false, b0.v,
                                                           (short)0, acc0, false, false);
            acc1 = __builtin_amdgcn_wmma_f32_16x16x32_bf16(false, a.v, false, b1.v,
                                                           (short)0, acc1, false, false);
        }
#pragma unroll
        for (int r = 0; r < 8; ++r) {
            int m = half * 8 + r;
            int gm = m0 + m;
            if (gm < N) {
                float v0 = acc0[r] + bias0; v0 = v0 > 0.0f ? v0 : 0.0f;
                float v1 = acc1[r] + bias1; v1 = v1 > 0.0f ? v1 : 0.0f;
                hout[(size_t)gm * DHID + n0 + nl] = v0;
                hout[(size_t)gm * DHID + n1 + nl] = v1;
            }
        }
    }
}

// ---------------------- graph mean-pool: accumulate ------------------------
__global__ __launch_bounds__(256) void pool_kernel(const float* __restrict__ h,
                                                   const int* __restrict__ batch,
                                                   float* __restrict__ sums,
                                                   float* __restrict__ cnt, int N) {
    long long t = (long long)blockIdx.x * 256 + threadIdx.x;
    const int perN = DHID >> 2;
    long long total = (long long)N * perN;
    if (t >= total) return;
    int n = (int)(t / perN);
    int f = ((int)(t - (long long)n * perN)) << 2;
    int g = batch[n];
    const float4 v = *(const float4*)&h[(size_t)n * DHID + f];
    float* o = &sums[(size_t)g * DHID + f];
    atomic_add_f32(o + 0, v.x);
    atomic_add_f32(o + 1, v.y);
    atomic_add_f32(o + 2, v.z);
    atomic_add_f32(o + 3, v.w);
    if (f == 0) atomic_add_f32(&cnt[g], 1.0f);
}

__global__ __launch_bounds__(256) void finalize_kernel(const float* __restrict__ sums,
                                                       const float* __restrict__ cnt,
                                                       float* __restrict__ out, int total) {
    int i = blockIdx.x * 256 + threadIdx.x;
    if (i >= total) return;
    float c = cnt[i / DHID];
    out[i] = sums[i] / (c > 1.0f ? c : 1.0f);
}

// ---------------------------------------------------------------------------
extern "C" void kernel_launch(void* const* d_in, const int* in_sizes, int n_in,
                              void* d_out, int out_size, void* d_ws, size_t ws_size,
                              hipStream_t stream) {
    const float* x     = (const float*)d_in[0];
    const int*   ei    = (const int*)d_in[1];
    const int*   batch = (const int*)d_in[2];
    const float* W1a   = (const float*)d_in[3];
    const float* b1a   = (const float*)d_in[4];
    const float* W1b   = (const float*)d_in[5];
    const float* b1b   = (const float*)d_in[6];
    const float* W2a   = (const float*)d_in[7];
    const float* b2a   = (const float*)d_in[8];
    const float* W2b   = (const float*)d_in[9];
    const float* b2b   = (const float*)d_in[10];

    const int N   = in_sizes[2];          // 50000
    const int E   = in_sizes[1] / 2;      // 800000
    const int DIN = in_sizes[0] / N;      // 128
    const int DH  = DHID;                 // 256
    const int*   src = ei;
    const int*   dst = ei + E;

    // ---- workspace layout ----
    float* agg  = (float*)d_ws;                                // N*256 f32
    float* hbuf = agg + (size_t)N * DH;                        // N*256 f32
    unsigned short* w1aT = (unsigned short*)(hbuf + (size_t)N * DH);
    unsigned short* w1bT = w1aT + (size_t)DIN * DH;
    unsigned short* w2aT = w1bT + (size_t)DH * DH;
    unsigned short* w2bT = w2aT + (size_t)DH * DH;
    float* pool = (float*)(w2bT + (size_t)DH * DH);            // 128*256 f32
    float* cnt  = pool + 128 * DH;                             // 128 f32

    // ---- convert weights to bf16, N-major ----
    {
        int tot1 = DIN * DH, tot2 = DH * DH;
        convert_w_kernel<<<(tot1 + 255) / 256, 256, 0, stream>>>(W1a, w1aT, DIN, DH);
        convert_w_kernel<<<(tot2 + 255) / 256, 256, 0, stream>>>(W1b, w1bT, DH, DH);
        convert_w_kernel<<<(tot2 + 255) / 256, 256, 0, stream>>>(W2a, w2aT, DH, DH);
        convert_w_kernel<<<(tot2 + 255) / 256, 256, 0, stream>>>(W2b, w2bT, DH, DH);
    }

    const int mlpBlocks = (N + 15) / 16;

    // ---- conv1 ----
    {
        long long nz = (long long)N * DIN;
        zero_kernel<<<(unsigned)((nz + 255) / 256), 256, 0, stream>>>(agg, nz);
        long long tot = (long long)E * (DIN / 4);
        scatter_kernel<<<(unsigned)((tot + 255) / 256), 256, 0, stream>>>(x, src, dst, agg, E, DIN);
        gin_mlp_kernel<<<mlpBlocks, 256, 0, stream>>>(x, agg, w1aT, b1a, w1bT, b1b, hbuf, N, DIN);
    }
    // ---- conv2 (in-place on hbuf: each block stages its rows before writing) ----
    {
        long long nz = (long long)N * DH;
        zero_kernel<<<(unsigned)((nz + 255) / 256), 256, 0, stream>>>(agg, nz);
        long long tot = (long long)E * (DH / 4);
        scatter_kernel<<<(unsigned)((tot + 255) / 256), 256, 0, stream>>>(hbuf, src, dst, agg, E, DH);
        gin_mlp_kernel<<<mlpBlocks, 256, 0, stream>>>(hbuf, agg, w2aT, b2a, w2bT, b2b, hbuf, N, DH);
    }
    // ---- global mean pool ----
    {
        long long nz = 128 * DH + 128;   // sums + counts (contiguous)
        zero_kernel<<<(unsigned)((nz + 255) / 256), 256, 0, stream>>>(pool, nz);
        long long tot = (long long)N * (DH / 4);
        pool_kernel<<<(unsigned)((tot + 255) / 256), 256, 0, stream>>>(hbuf, batch, pool, cnt, N);
        int totOut = 128 * DH;
        finalize_kernel<<<(totOut + 255) / 256, 256, 0, stream>>>(pool, cnt, (float*)d_out, totOut);
    }
}
